// TemporalTopKAttention_59596966199851
// MI455X (gfx1250) — matrix operand order repaired and checked by reading, not compiled, not run on hardware
//
#include <hip/hip_runtime.h>
#include <hip/hip_bf16.h>

// ---------------------------------------------------------------------------
// Temporal top-k block-sparse attention for MI455X (gfx1250, wave32, WMMA).
// GEMMs: v_wmma_f32_16x16x32_bf16, TDM (tensor_load_to_lds) A-panel staging,
// b128 global loads for transposed-weight B tiles.
// ---------------------------------------------------------------------------

typedef __attribute__((ext_vector_type(16))) __bf16 v16bf;
typedef __attribute__((ext_vector_type(8)))  float  v8f;
typedef __attribute__((ext_vector_type(4)))  unsigned int v4u;
typedef __attribute__((ext_vector_type(8)))  int v8i;
typedef __attribute__((ext_vector_type(4)))  int v4i;

#if defined(__has_builtin)
#  if __has_builtin(__builtin_amdgcn_tensor_load_to_lds)
#    define HAVE_TDM 1
#  endif
#endif
#ifndef HAVE_TDM
#  define HAVE_TDM 0
#endif

namespace {
constexpr int kB   = 4;
constexpr int kT   = 2048;
constexpr int kC   = 512;
constexpr int kH   = 8;
constexpr int kDh  = 64;
constexpr int kR   = 16;
constexpr int kNr  = kT / kR;   // 128 regions
constexpr int kK   = 8;         // top-k regions
constexpr int k3C  = 3 * kC;    // 1536
constexpr float kScale = 0.125f; // 1/sqrt(Dh)
}

static __device__ inline v8f wmma_bf16(v16bf a, v16bf b, v8f c) {
  return __builtin_amdgcn_wmma_f32_16x16x32_bf16(false, a, false, b,
                                                 (short)0, c, false, false);
}

union BfPack { v16bf v; v4u q[2]; };

// A-matrix (16x32 bf16): lane m=l&15; per lane K = hi*16 + 0..15 contiguous
// -> 2x b128.
static __device__ inline v16bf load_A16x32(const __bf16* A, int lda,
                                           int row0, int k0, int lane) {
  const int m  = lane & 15;
  const int hi = (lane >> 4) & 1;
  const __bf16* base = A + (size_t)(row0 + m) * lda + k0 + hi * 16;
  BfPack u;
  u.q[0] = *(const v4u*)(base);
  u.q[1] = *(const v4u*)(base + 8);
  return u.v;
}

// B-matrix (32x16 bf16) from a K-contiguous source: lane n=l&15 is the
// column, per lane K = hi*16 + 0..15 contiguous -> 2x b128.
static __device__ inline v16bf load_B32x16_kcontig(const __bf16* src, int ldk,
                                                   int k0, int col0, int lane) {
  const int n  = lane & 15;
  const int hi = (lane >> 4) & 1;
  const __bf16* base = src + (size_t)(col0 + n) * ldk + k0 + hi * 16;
  BfPack u;
  u.q[0] = *(const v4u*)(base);
  u.q[1] = *(const v4u*)(base + 8);
  return u.v;
}

// ---------------------------------------------------------------------------
// TDM: DMA a 2-D bf16 tile (height rows x width elems, row stride in elems)
// from global memory into LDS.  D# packed per CDNA5 ISA §8.3/8.4.
// Call from ONE wave; follows with s_wait_tensorcnt 0.
// ---------------------------------------------------------------------------
static __device__ inline void tdm_load_2d(__bf16* lds_ptr,
                                          const __bf16* gptr,
                                          int width, int height,
                                          int stride_elems, int lane) {
#if HAVE_TDM
  const unsigned long long ga = (unsigned long long)(uintptr_t)gptr;
  const unsigned lds_off = (unsigned)(uintptr_t)lds_ptr;  // low 32b = LDS byte addr
  v4u g0;
  g0[0] = 1u;                                   // count=1 (valid user D#)
  g0[1] = lds_off;                              // lds_addr
  g0[2] = (unsigned)(ga & 0xffffffffu);         // global_addr[31:0]
  g0[3] = (unsigned)((ga >> 32) & 0x01ffffffu)  // global_addr[56:32]
          | (2u << 30);                         // type = 2 ("image")
  v8i g1;
  g1[0] = 1 << 16;                              // data_size = 1 -> 2 bytes
  g1[1] = (int)(((unsigned)width & 0xffffu) << 16);          // tensor_dim0 lo
  g1[2] = (int)((((unsigned)width >> 16) & 0xffffu)
          | (((unsigned)height & 0xffffu) << 16));           // dim0 hi | dim1 lo
  g1[3] = (int)((((unsigned)height >> 16) & 0xffffu)
          | (((unsigned)width & 0xffffu) << 16));            // dim1 hi | tile_dim0
  g1[4] = (int)((unsigned)height & 0xffffu);                 // tile_dim1 (tile_dim2=0)
  g1[5] = (int)(unsigned)stride_elems;                       // dim0_stride lo32
  g1[6] = 0;                                                 // stride hi | dim1_stride lo
  g1[7] = 0;
  v4i g2 = {0, 0, 0, 0};
  v4i g3 = {0, 0, 0, 0};
  v8i g4 = {0, 0, 0, 0, 0, 0, 0, 0};
  __builtin_amdgcn_tensor_load_to_lds(g0, g1, g2, g3, g4, 0);
  __builtin_amdgcn_s_wait_tensorcnt(0);
#else
  const int total = width * height;
  for (int i = lane; i < total; i += 32) {
    const int r = i / width, c = i - r * width;
    lds_ptr[i] = gptr[(size_t)r * stride_elems + c];
  }
#endif
}

// ---------------------------------------------------------------------------
// 1) fp32 -> bf16 conversion (straight) and transposing variant for weights
// ---------------------------------------------------------------------------
__global__ void cvt_f32_bf16_kernel(const float* __restrict__ in,
                                    __bf16* __restrict__ out, int n) {
  int i = blockIdx.x * blockDim.x + threadIdx.x;
  int stride = gridDim.x * blockDim.x;
  for (; i < n; i += stride) out[i] = (__bf16)in[i];
}

// W stored (K,N) row-major fp32 -> W^T (N,K) bf16, K contiguous.
__global__ void cvt_transpose_kernel(const float* __restrict__ in,
                                     __bf16* __restrict__ out, int N) {
  const int idx = blockIdx.x * blockDim.x + threadIdx.x;  // idx = n*kC + k
  if (idx >= N * kC) return;
  const int k = idx & (kC - 1);
  const int n = idx >> 9;  // / kC
  out[idx] = (__bf16)in[(size_t)k * N + n];
}

// ---------------------------------------------------------------------------
// 2) QKV projection: (B*T,512) x (512,1536).  Block = 8 waves = 128x64 tile.
//    A panel (128x32) staged via TDM into LDS each K-step; B from W^T b128.
// ---------------------------------------------------------------------------
__global__ __launch_bounds__(256) void qkv_gemm_kernel(
    const __bf16* __restrict__ xb, const __bf16* __restrict__ wT,
    const float* __restrict__ bias, __bf16* __restrict__ qo,
    __bf16* __restrict__ ko, __bf16* __restrict__ vo) {
  __shared__ __bf16 smemA[128 * 32];  // 8 KB
  const int lane = threadIdx.x & 31;
  const int w    = threadIdx.x >> 5;          // wave id, 0..7
  const int rowB = blockIdx.x * 128;          // M block
  const int col0 = blockIdx.y * 64;           // N block

  v8f acc[4];
#pragma unroll
  for (int t = 0; t < 4; ++t)
#pragma unroll
    for (int i = 0; i < 8; ++i) acc[t][i] = 0.0f;

  for (int k0 = 0; k0 < kC; k0 += 32) {
    __syncthreads();  // previous chunk fully consumed
    if (threadIdx.x < 32) {
      tdm_load_2d(smemA, xb + (size_t)rowB * kC + k0, /*w=*/32, /*h=*/128,
                  /*stride=*/kC, lane);
    }
    __syncthreads();  // LDS chunk visible to all waves

    if (k0 + 32 < kC)
      __builtin_prefetch(wT + (size_t)(col0 + (lane & 15)) * kC + k0 + 32, 0, 1);

    // A tile for this wave from LDS (ds_load_b128 x2)
    const v16bf a = load_A16x32(smemA, 32, w * 16, 0, lane);
#pragma unroll
    for (int t = 0; t < 4; ++t) {
      const v16bf b = load_B32x16_kcontig(wT, kC, k0, col0 + 16 * t, lane);
      acc[t] = wmma_bf16(a, b, acc[t]);
    }
  }

  // Epilogue: bias add + scatter to (B,H,T,Dh) q/k/v bf16 buffers.
  const int n  = lane & 15;
  const int hi = (lane >> 4) & 1;
#pragma unroll
  for (int t = 0; t < 4; ++t) {
    const int ncol = col0 + 16 * t + n;
    const int part = ncol >> 9;       // 0=q 1=k 2=v
    const int c    = ncol & (kC - 1);
    const int h    = c >> 6;
    const int d    = c & (kDh - 1);
    __bf16* dst = (part == 0) ? qo : (part == 1) ? ko : vo;
    const float bv = bias[ncol];
#pragma unroll
    for (int r = 0; r < 8; ++r) {
      const int row = rowB + w * 16 + hi * 8 + r;
      const int b   = row >> 11;          // / T
      const int tt  = row & (kT - 1);
      const float val = acc[t][r] + bv;
      dst[(((size_t)b * kH + h) * kT + tt) * kDh + d] = (__bf16)val;
    }
  }
}

// ---------------------------------------------------------------------------
// 3) Region means of q and k: (B,H,Nr,Dh)
// ---------------------------------------------------------------------------
__global__ void region_avg_kernel(const __bf16* __restrict__ q,
                                  const __bf16* __restrict__ k,
                                  float* __restrict__ qa,
                                  float* __restrict__ ka) {
  const int total = kB * kH * kNr * kDh;
  int idx = blockIdx.x * blockDim.x + threadIdx.x;
  if (idx >= total) return;
  const int d  = idx & (kDh - 1);
  const int nr = (idx >> 6) & (kNr - 1);
  const int bh = idx >> 13;
  const size_t base = ((size_t)bh * kT + nr * kR) * kDh + d;
  float sq = 0.0f, sk = 0.0f;
#pragma unroll
  for (int r = 0; r < kR; ++r) {
    sq += (float)q[base + (size_t)r * kDh];
    sk += (float)k[base + (size_t)r * kDh];
  }
  qa[idx] = sq * (1.0f / kR);
  ka[idx] = sk * (1.0f / kR);
}

// ---------------------------------------------------------------------------
// 4) Region scores + top-8. One wave per (b,h,query-region).
// ---------------------------------------------------------------------------
__global__ __launch_bounds__(32) void topk_kernel(
    const float* __restrict__ qa, const float* __restrict__ ka,
    int* __restrict__ topk) {
  const int bhq  = blockIdx.x;            // B*H*Nr
  const int lane = threadIdx.x & 31;
  const int bh   = bhq >> 7;
  const int qnr  = bhq & (kNr - 1);
  const float* qv = qa + ((size_t)bh * kNr + qnr) * kDh;

  float vals[4];
  int   idxs[4];
#pragma unroll
  for (int i = 0; i < 4; ++i) {
    const int j = lane + 32 * i;
    const float* kv = ka + ((size_t)bh * kNr + j) * kDh;
    float s = 0.0f;
    for (int d = 0; d < kDh; ++d) s += qv[d] * kv[d];
    vals[i] = s * kScale;
    idxs[i] = j;
  }

  const int out_base = bhq * kK;
#pragma unroll
  for (int t = 0; t < kK; ++t) {
    float m  = vals[0];
    int   mj = idxs[0];
#pragma unroll
    for (int i = 1; i < 4; ++i)
      if (vals[i] > m) { m = vals[i]; mj = idxs[i]; }
    for (int off = 16; off > 0; off >>= 1) {
      const float om = __shfl_xor(m, off);
      const int   oj = __shfl_xor(mj, off);
      if (om > m || (om == m && oj < mj)) { m = om; mj = oj; }
    }
    if (lane == 0) topk[out_base + t] = mj;
#pragma unroll
    for (int i = 0; i < 4; ++i)
      if (idxs[i] == mj) vals[i] = -3.0e38f;
  }
}

// ---------------------------------------------------------------------------
// 5) Block-sparse attention. One wave per (b,h,query-region).
// ---------------------------------------------------------------------------
__global__ __launch_bounds__(32) void attn_kernel(
    const __bf16* __restrict__ q, const __bf16* __restrict__ k,
    const __bf16* __restrict__ v, const int* __restrict__ topk,
    __bf16* __restrict__ y) {
  const int bhq  = blockIdx.x;
  const int lane = threadIdx.x & 31;
  const int bh   = bhq >> 7;
  const int qnr  = bhq & (kNr - 1);
  const int b    = bh >> 3;
  const int h    = bh & (kH - 1);

  const __bf16* qp = q + (size_t)bh * kT * kDh;
  const __bf16* kp = k + (size_t)bh * kT * kDh;
  const __bf16* vp = v + (size_t)bh * kT * kDh;

  int regs[kK];
#pragma unroll
  for (int t = 0; t < kK; ++t) regs[t] = topk[bhq * kK + t];

  const int n  = lane & 15;
  const int hi = (lane >> 4) & 1;

  const v16bf a0 = load_A16x32(qp, kDh, qnr * kR, 0, lane);
  const v16bf a1 = load_A16x32(qp, kDh, qnr * kR, 32, lane);

  v8f acc[8];
#pragma unroll
  for (int jt = 0; jt < 8; ++jt)
#pragma unroll
    for (int i = 0; i < 8; ++i) acc[jt][i] = 0.0f;

  // logits: B-tile column j = gathered key row; K=d contiguous -> b128 loads
#pragma unroll
  for (int jt = 0; jt < 8; ++jt) {
    const int trow = regs[jt] * kR + n;
    const __bf16* kbase = kp + (size_t)trow * kDh + hi * 16;
    BfPack b0, b1;
    b0.q[0] = *(const v4u*)(kbase);
    b0.q[1] = *(const v4u*)(kbase + 8);
    b1.q[0] = *(const v4u*)(kbase + 32);
    b1.q[1] = *(const v4u*)(kbase + 40);
    acc[jt] = wmma_bf16(a0, b0.v, acc[jt]);
    acc[jt] = wmma_bf16(a1, b1.v, acc[jt]);
  }

  // softmax per row (row m = r + 8*hi across 16-lane halves)
#pragma unroll
  for (int r = 0; r < 8; ++r) {
    float mx = -3.0e38f;
#pragma unroll
    for (int jt = 0; jt < 8; ++jt) {
      acc[jt][r] *= kScale;
      mx = fmaxf(mx, acc[jt][r]);
    }
    for (int off = 8; off > 0; off >>= 1) mx = fmaxf(mx, __shfl_xor(mx, off));
    float sum = 0.0f;
#pragma unroll
    for (int jt = 0; jt < 8; ++jt) {
      const float e = __expf(acc[jt][r] - mx);
      acc[jt][r] = e;
      sum += e;
    }
    for (int off = 8; off > 0; off >>= 1) sum += __shfl_xor(sum, off);
    const float inv = 1.0f / sum;
#pragma unroll
    for (int jt = 0; jt < 8; ++jt) acc[jt][r] *= inv;
  }

  // stage attn (16x128) to LDS bf16 (C-layout -> A-layout transpose)
  __shared__ __bf16 smem[16 * 136];
#pragma unroll
  for (int jt = 0; jt < 8; ++jt)
#pragma unroll
    for (int r = 0; r < 8; ++r)
      smem[(r + 8 * hi) * 136 + jt * 16 + n] = (__bf16)acc[jt][r];
  __syncthreads();

  v8f oacc[4];
#pragma unroll
  for (int dt = 0; dt < 4; ++dt)
#pragma unroll
    for (int i = 0; i < 8; ++i) oacc[dt][i] = 0.0f;

#pragma unroll
  for (int kt = 0; kt < 4; ++kt) {
    const int ks = kt * 32;
    const v16bf aA = load_A16x32(smem, 136, 0, ks, lane);
#pragma unroll
    for (int dt = 0; dt < 4; ++dt) {
      v16bf bv;
#pragma unroll
      for (int vv = 0; vv < 8; ++vv) {
        const int j0 = ks + hi * 16 + 2 * vv;
        const int j1 = j0 + 1;
        const int t0 = regs[j0 >> 4] * kR + (j0 & 15);
        const int t1 = regs[j1 >> 4] * kR + (j1 & 15);
        bv[2 * vv]     = vp[(size_t)t0 * kDh + dt * 16 + n];
        bv[2 * vv + 1] = vp[(size_t)t1 * kDh + dt * 16 + n];
      }
      oacc[dt] = wmma_bf16(aA, bv, oacc[dt]);
    }
  }

#pragma unroll
  for (int dt = 0; dt < 4; ++dt) {
    const int d = dt * 16 + n;
#pragma unroll
    for (int r = 0; r < 8; ++r) {
      const int m = r + 8 * hi;
      const int t = qnr * kR + m;
      y[((size_t)b * kT + t) * kC + h * kDh + d] = (__bf16)oacc[dt][r];
    }
  }
}

// ---------------------------------------------------------------------------
// 6) Output projection: (B*T,512) x (512,512) + bout -> f32 d_out
// ---------------------------------------------------------------------------
__global__ __launch_bounds__(256) void out_gemm_kernel(
    const __bf16* __restrict__ yb, const __bf16* __restrict__ wT,
    const float* __restrict__ bias, float* __restrict__ out) {
  __shared__ __bf16 smemA[128 * 32];
  const int lane = threadIdx.x & 31;
  const int w    = threadIdx.x >> 5;
  const int rowB = blockIdx.x * 128;
  const int col0 = blockIdx.y * 64;

  v8f acc[4];
#pragma unroll
  for (int t = 0; t < 4; ++t)
#pragma unroll
    for (int i = 0; i < 8; ++i) acc[t][i] = 0.0f;

  for (int k0 = 0; k0 < kC; k0 += 32) {
    __syncthreads();
    if (threadIdx.x < 32) {
      tdm_load_2d(smemA, yb + (size_t)rowB * kC + k0, 32, 128, kC, lane);
    }
    __syncthreads();

    if (k0 + 32 < kC)
      __builtin_prefetch(wT + (size_t)(col0 + (lane & 15)) * kC + k0 + 32, 0, 1);

    const v16bf a = load_A16x32(smemA, 32, w * 16, 0, lane);
#pragma unroll
    for (int t = 0; t < 4; ++t) {
      const v16bf b = load_B32x16_kcontig(wT, kC, k0, col0 + 16 * t, lane);
      acc[t] = wmma_bf16(a, b, acc[t]);
    }
  }

  const int n  = lane & 15;
  const int hi = (lane >> 4) & 1;
#pragma unroll
  for (int t = 0; t < 4; ++t) {
    const int ncol = col0 + 16 * t + n;
    const float bv = bias[ncol];
#pragma unroll
    for (int r = 0; r < 8; ++r) {
      const int row = rowB + w * 16 + hi * 8 + r;
      out[(size_t)row * kC + ncol] = acc[t][r] + bv;
    }
  }
}

// ---------------------------------------------------------------------------
// Launch
// ---------------------------------------------------------------------------
extern "C" void kernel_launch(void* const* d_in, const int* in_sizes, int n_in,
                              void* d_out, int out_size, void* d_ws,
                              size_t ws_size, hipStream_t stream) {
  const float* x    = (const float*)d_in[0];
  const float* Wqkv = (const float*)d_in[1];
  const float* bqkv = (const float*)d_in[2];
  const float* Wout = (const float*)d_in[3];
  const float* bout = (const float*)d_in[4];
  float* out = (float*)d_out;

  char* ws = (char*)d_ws;
  size_t off = 0;
  auto carve = [&](size_t bytes) -> void* {
    void* p = ws + off;
    off = (off + bytes + 255) & ~(size_t)255;
    return p;
  };

  __bf16* xb     = (__bf16*)carve((size_t)kB * kT * kC * sizeof(__bf16));
  __bf16* wqkvT  = (__bf16*)carve((size_t)kC * k3C * sizeof(__bf16));
  __bf16* woutT  = (__bf16*)carve((size_t)kC * kC * sizeof(__bf16));
  __bf16* qb     = (__bf16*)carve((size_t)kB * kH * kT * kDh * sizeof(__bf16));
  __bf16* kb     = (__bf16*)carve((size_t)kB * kH * kT * kDh * sizeof(__bf16));
  __bf16* vb     = (__bf16*)carve((size_t)kB * kH * kT * kDh * sizeof(__bf16));
  float*  qa     = (float*)carve((size_t)kB * kH * kNr * kDh * sizeof(float));
  float*  ka     = (float*)carve((size_t)kB * kH * kNr * kDh * sizeof(float));
  int*    tk     = (int*)carve((size_t)kB * kH * kNr * kK * sizeof(int));
  __bf16* yb     = (__bf16*)carve((size_t)kB * kT * kC * sizeof(__bf16));

  const int nx = kB * kT * kC;
  cvt_f32_bf16_kernel<<<(nx + 255) / 256, 256, 0, stream>>>(x, xb, nx);

  const int nw1 = kC * k3C;  // transposed: (N=1536) x (K=512)
  cvt_transpose_kernel<<<(nw1 + 255) / 256, 256, 0, stream>>>(Wqkv, wqkvT, k3C);
  const int nw2 = kC * kC;
  cvt_transpose_kernel<<<(nw2 + 255) / 256, 256, 0, stream>>>(Wout, woutT, kC);

  qkv_gemm_kernel<<<dim3(kB * kT / 128, k3C / 64), 256, 0, stream>>>(
      xb, wqkvT, bqkv, qb, kb, vb);

  const int na = kB * kH * kNr * kDh;
  region_avg_kernel<<<(na + 255) / 256, 256, 0, stream>>>(qb, kb, qa, ka);

  topk_kernel<<<kB * kH * kNr, 32, 0, stream>>>(qa, ka, tk);

  attn_kernel<<<kB * kH * kNr, 32, 0, stream>>>(qb, kb, vb, tk, yb);

  out_gemm_kernel<<<dim3(kB * kT / 128, kC / 64), 256, 0, stream>>>(
      yb, woutT, bout, out);
}
